// LocalAttention_45629732553222
// MI455X (gfx1250) — matrix operand order repaired and checked by reading, not compiled, not run on hardware
//
#include <hip/hip_runtime.h>

// MI455X / gfx1250, wave32. bf16 WMMA (f32 accum) for all three GEMM stages,
// async global->LDS staging (ASYNCcnt) for the attention operand tiles.
// Workspace (~64.4 MB):
//   theta bf16[6422528] | phi bf16[6422528] | g bf16[6422528] | folded f32[6422528] | sums f32[128]

typedef __bf16 bf16_t;
typedef __attribute__((ext_vector_type(16))) __bf16 v16bf;
typedef __attribute__((ext_vector_type(8)))  __bf16 v8bf;
typedef __attribute__((ext_vector_type(8)))  float  v8f;

#define B_    8
#define C_    64
#define T_    4
#define H_    56
#define W_    56
#define HW_   3136
#define N1_   100352        // B*T*HW
#define HID_  64
#define TK2_  36
#define NELEM 6422528       // C * N1
#define STR   72            // LDS row stride (bf16), 144B rows, 16B-aligned

__device__ __forceinline__ v8f wmma_bf16(v16bf a, v16bf b, v8f c) {
  return __builtin_amdgcn_wmma_f32_16x16x32_bf16(false, a, false, b, (short)0, c, false, false);
}

// A-matrix fragment (16x32 bf16): elem e -> K = kb + kh*16 + e (contiguous run)
__device__ __forceinline__ v16bf ldsA(const bf16_t* row, int kh, int kb) {
  v8bf lo = *(const v8bf*)(row + kb + kh * 16);
  v8bf hi = *(const v8bf*)(row + kb + kh * 16 + 8);
  v16bf r;
#pragma unroll
  for (int e = 0; e < 8; ++e) { r[e] = lo[e]; r[8 + e] = hi[e]; }
  return r;
}
// B-matrix fragment (32x16 bf16): elem e -> K = kb + kh*16 + e (contiguous run)
__device__ __forceinline__ v16bf ldsB(const bf16_t* row, int kh, int kb) {
  v8bf lo = *(const v8bf*)(row + kb + kh * 16);
  v8bf hi = *(const v8bf*)(row + kb + kh * 16 + 8);
  v16bf r;
#pragma unroll
  for (int e = 0; e < 8; ++e) { r[e] = lo[e]; r[8 + e] = hi[e]; }
  return r;
}

// ---------------- conv_in: h = w_in(192x64) @ X(64xN1) -> theta/phi/g (bf16, c-contiguous).
// One wave per N-tile; B-fragments loaded once, reused across all 12 M-tiles.
__global__ __launch_bounds__(256) void conv_in_kernel(
    const float* __restrict__ x, const float* __restrict__ w_in,
    const float* __restrict__ b_in,
    bf16_t* __restrict__ theta, bf16_t* __restrict__ phi, bf16_t* __restrict__ gbuf) {
  int nt   = blockIdx.x * 8 + (threadIdx.x >> 5);          // 0..6271
  int lane = threadIdx.x & 31, kh = lane >> 4, ln = lane & 15;
  int n  = nt * 16 + ln;
  int b  = n / (T_ * HW_);
  int rem = n % (T_ * HW_);
  int t  = rem / HW_;
  int hw = rem % HW_;
  const float* xcol = x + (size_t)((b * C_) * T_ + t) * HW_ + hw;   // + c*T_*HW_
  v16bf b0, b1;
#pragma unroll
  for (int e = 0; e < 16; ++e) {
    b0[e] = (bf16_t)xcol[(size_t)(kh * 16 + e) * (T_ * HW_)];
    b1[e] = (bf16_t)xcol[(size_t)(32 + kh * 16 + e) * (T_ * HW_)];
  }
  size_t dofs = (size_t)((b * T_ + t) * HW_ + hw) * 64;
#pragma unroll
  for (int mt = 0; mt < 12; ++mt) {
    const float* wrow = w_in + (mt * 16 + ln) * C_;
    v16bf a0, a1;
#pragma unroll
    for (int e = 0; e < 16; ++e) {
      int ka = kh * 16 + e;
      a0[e] = (bf16_t)wrow[ka];
      a1[e] = (bf16_t)wrow[32 + ka];
    }
    v8f acc = {};
    acc = wmma_bf16(a0, b0, acc);
    acc = wmma_bf16(a1, b1, acc);
    int sel = mt >> 2;
    bf16_t* dst = (sel == 0 ? theta : (sel == 1 ? phi : gbuf)) + dofs;
#pragma unroll
    for (int r = 0; r < 8; ++r) {
      int m = mt * 16 + r + 8 * kh;
      dst[m & 63] = (bf16_t)(acc[r] + b_in[m]);
    }
  }
}

// ---------------- per-site attention: attn = tu^T pu / 36 ; out = gu attn^T ; fold via atomics.
// theta/phi tiles staged with global_load_async_to_lds_b128 (ASYNCcnt), g transposed in LDS.
__global__ __launch_bounds__(64) void attn_kernel(
    const bf16_t* __restrict__ theta, const bf16_t* __restrict__ phi,
    const bf16_t* __restrict__ gbuf, float* __restrict__ folded) {
  __shared__ __align__(16) bf16_t lds[2][208 * STR];
  int wid  = threadIdx.x >> 5;
  int lane = threadIdx.x & 31, kh = lane >> 4, ln = lane & 15;
  int site = blockIdx.x * 2 + wid;
  int b = site / HW_, hw = site % HW_;
  int h = hw / W_,   w  = hw % W_;
  bf16_t* th = lds[wid];          // [t1][c]  (matmul1 A)
  bf16_t* ph = th + 48 * STR;     // [t2][c]  (matmul1 B)
  bf16_t* gt = ph + 48 * STR;     // [c][t2]  (matmul2 A, transposed)
  bf16_t* at = gt + 64 * STR;     // [t1][t2] (attn; matmul2 B)
  unsigned th_l = (unsigned)(uintptr_t)th;
  unsigned ph_l = (unsigned)(uintptr_t)ph;
  const v8bf z8 = {};

  // Stage theta/phi rows: 48 rows x 8 chunks (16B) per matrix.
  // In-bounds rows -> async global->LDS copy; halo / pad rows -> zero ds_store.
  for (int i = lane; i < 2 * 48 * 8; i += 32) {
    int mat = i / 384;
    int i2  = i % 384;
    int row = i2 >> 3, ch = i2 & 7;
    int t = row / 9, idx = row % 9;
    int y = h + idx / 3 - 1, xx = w + idx % 3 - 1;
    bool inb = (row < TK2_) && y >= 0 && y < H_ && xx >= 0 && xx < W_;
    if (inb) {
      size_t gi = (size_t)((b * T_ + t) * HW_ + y * W_ + xx) * 64 + ch * 8;
      const bf16_t* gp = (mat ? phi : theta) + gi;
      unsigned la = (mat ? ph_l : th_l) + row * (STR * 2) + ch * 16;
      asm volatile("global_load_async_to_lds_b128 %0, %1, off"
                   :: "v"(la), "v"(gp) : "memory");
    } else {
      *(v8bf*)((mat ? ph : th) + row * STR + ch * 8) = z8;
    }
  }
  // Overlap with in-flight asyncs: zero K-pad columns of gt (t2 >= 36) and at (t2 >= 48).
  for (int i = lane; i < 64 * 32; i += 32) {      // gt rows 0..63, cols 36..67
    gt[(i >> 5) * STR + 36 + (i & 31)] = (bf16_t)0.f;
  }
  for (int i = lane; i < 48 * 16; i += 32) {      // at rows 0..47, cols 48..63
    at[(i >> 4) * STR + 48 + (i & 15)] = (bf16_t)0.f;
  }
  // g transposed fill: gt[c][t2] (global reads overlap asyncs too).
  for (int i = lane; i < TK2_ * 64; i += 32) {
    int row = i >> 6, c = i & 63;
    int t = row / 9, idx = row % 9;
    int y = h + idx / 3 - 1, xx = w + idx % 3 - 1;
    bf16_t gv = (bf16_t)0.f;
    if (y >= 0 && y < H_ && xx >= 0 && xx < W_)
      gv = gbuf[(size_t)((b * T_ + t) * HW_ + y * W_ + xx) * 64 + c];
    gt[c * STR + row] = gv;
  }
  asm volatile("s_wait_asynccnt 0x0" ::: "memory");
  __syncthreads();

  // matmul1: attn[t1][t2] = sum_c th[t1][c]*ph[t2][c]  (M=48,N=48,K=64)
#pragma unroll
  for (int mt = 0; mt < 3; ++mt) {
    const bf16_t* arow = th + (mt * 16 + ln) * STR;
    v16bf a0 = ldsA(arow, kh, 0), a1 = ldsA(arow, kh, 32);
#pragma unroll
    for (int nt = 0; nt < 3; ++nt) {
      const bf16_t* brow = ph + (nt * 16 + ln) * STR;
      v8f d = {};
      d = wmma_bf16(a0, ldsB(brow, kh, 0), d);
      d = wmma_bf16(a1, ldsB(brow, kh, 32), d);
#pragma unroll
      for (int r = 0; r < 8; ++r)
        at[(mt * 16 + r + 8 * kh) * STR + nt * 16 + ln] = (bf16_t)(d[r] * (1.f / 36.f));
    }
  }
  __syncthreads();

  // matmul2: out[c][t1] = sum_t2 gt[c][t2]*at[t1][t2]  (M=64,N=48,K=64 zero-padded),
  // folded directly via f32 atomics (8 consecutive channels per lane).
#pragma unroll
  for (int mt = 0; mt < 4; ++mt) {
    const bf16_t* arow = gt + (mt * 16 + ln) * STR;
    v16bf a0 = ldsA(arow, kh, 0), a1 = ldsA(arow, kh, 32);
#pragma unroll
    for (int nt = 0; nt < 3; ++nt) {
      const bf16_t* brow = at + (nt * 16 + ln) * STR;
      v8f d = {};
      d = wmma_bf16(a0, ldsB(brow, kh, 0), d);
      d = wmma_bf16(a1, ldsB(brow, kh, 32), d);
      int t1i = nt * 16 + ln;                 // D col = lane&15 -> t1 index
      if (t1i < TK2_) {
        int t1 = t1i / 9, idx = t1i % 9;
        int y = h + idx / 3 - 1, xx = w + idx % 3 - 1;
        if (y >= 0 && y < H_ && xx >= 0 && xx < W_) {
          float* dst = folded + (size_t)((b * T_ + t1) * HW_ + y * W_ + xx) * 64 +
                       mt * 16 + 8 * kh;
#pragma unroll
          for (int r = 0; r < 8; ++r) atomicAdd(dst + r, d[r]);
        }
      }
    }
  }
}

// ---------------- conv_out: y = w_out(64x64) @ folded(64xN1) + b_out -> d_out (pre-BN).
// One wave per N-tile, B-fragments reused across the 4 M-tiles.
__global__ __launch_bounds__(256) void conv_out_kernel(
    const float* __restrict__ folded, const float* __restrict__ w_out,
    const float* __restrict__ b_out, float* __restrict__ y) {
  int nt   = blockIdx.x * 8 + (threadIdx.x >> 5);
  int lane = threadIdx.x & 31, kh = lane >> 4, ln = lane & 15;
  int n  = nt * 16 + ln;
  int b  = n / (T_ * HW_);
  int rem = n % (T_ * HW_);
  int t  = rem / HW_;
  int hw = rem % HW_;
  const float* fcol = folded + (size_t)n * 64;
  v16bf b0, b1;
#pragma unroll
  for (int e = 0; e < 16; ++e) {
    b0[e] = (bf16_t)fcol[kh * 16 + e];
    b1[e] = (bf16_t)fcol[32 + kh * 16 + e];
  }
#pragma unroll
  for (int mt = 0; mt < 4; ++mt) {
    const float* wrow = w_out + (mt * 16 + ln) * HID_;
    v16bf a0, a1;
#pragma unroll
    for (int e = 0; e < 16; ++e) {
      int ka = kh * 16 + e;
      a0[e] = (bf16_t)wrow[ka];
      a1[e] = (bf16_t)wrow[32 + ka];
    }
    v8f acc = {};
    acc = wmma_bf16(a0, b0, acc);
    acc = wmma_bf16(a1, b1, acc);
#pragma unroll
    for (int r = 0; r < 8; ++r) {
      int c = mt * 16 + r + 8 * kh;
      y[(size_t)((b * C_ + c) * T_ + t) * HW_ + hw] = acc[r] + b_out[c];
    }
  }
}

// ---------------- BN stats: per-(b,c) slab reduction -> sums[c], sums[64+c]
__global__ __launch_bounds__(256) void stats_kernel(const float* __restrict__ y,
                                                    float* __restrict__ sums) {
  __shared__ float s1[256], s2[256];
  int c = blockIdx.x & 63, b = blockIdx.x >> 6;
  const float* base = y + (size_t)((b * C_ + c) * T_) * HW_;
  float a1 = 0.f, a2 = 0.f;
  for (int i = threadIdx.x; i < T_ * HW_; i += 256) {
    float v = base[i]; a1 += v; a2 += v * v;
  }
  s1[threadIdx.x] = a1; s2[threadIdx.x] = a2;
  __syncthreads();
  for (int off = 128; off > 0; off >>= 1) {
    if ((int)threadIdx.x < off) {
      s1[threadIdx.x] += s1[threadIdx.x + off];
      s2[threadIdx.x] += s2[threadIdx.x + off];
    }
    __syncthreads();
  }
  if (threadIdx.x == 0) {
    atomicAdd(&sums[c], s1[0]);
    atomicAdd(&sums[64 + c], s2[0]);
  }
}

// ---------------- finalize: out = x + gamma*(y-mu)*rsqrt(var+eps) + beta (in place on d_out)
__global__ __launch_bounds__(256) void final_kernel(
    const float* __restrict__ x, const float* __restrict__ gamma,
    const float* __restrict__ beta, const float* __restrict__ sums,
    float* __restrict__ y) {
  int i = blockIdx.x * 256 + threadIdx.x;
  int c = (i / (T_ * HW_)) & 63;
  const float invN = 1.f / (float)N1_;
  float mu  = sums[c] * invN;
  float var = sums[64 + c] * invN - mu * mu;
  float v = (y[i] - mu) * rsqrtf(var + 1e-5f);
  y[i] = x[i] + v * gamma[c] + beta[c];
}

__global__ __launch_bounds__(256) void zero_kernel(float* __restrict__ p, int n) {
  int i = blockIdx.x * 256 + threadIdx.x;
  if (i < n) p[i] = 0.f;
}

extern "C" void kernel_launch(void* const* d_in, const int* in_sizes, int n_in,
                              void* d_out, int out_size, void* d_ws, size_t ws_size,
                              hipStream_t stream) {
  const float* x     = (const float*)d_in[0];
  const float* w_in  = (const float*)d_in[1];
  const float* b_in  = (const float*)d_in[2];
  const float* w_out = (const float*)d_in[3];
  const float* b_out = (const float*)d_in[4];
  const float* gamma = (const float*)d_in[5];
  const float* beta  = (const float*)d_in[6];
  float* out = (float*)d_out;

  bf16_t* theta  = (bf16_t*)d_ws;
  bf16_t* phi    = theta + NELEM;
  bf16_t* gbuf   = phi + NELEM;
  float*  folded = (float*)(gbuf + NELEM);
  float*  sums   = folded + NELEM;          // 128 floats, contiguous after folded

  zero_kernel<<<(NELEM + 128 + 255) / 256, 256, 0, stream>>>(folded, NELEM + 128);
  conv_in_kernel<<<6272 / 8, 256, 0, stream>>>(x, w_in, b_in, theta, phi, gbuf);
  attn_kernel<<<(B_ * HW_) / 2, 64, 0, stream>>>(theta, phi, gbuf, folded);
  conv_out_kernel<<<6272 / 8, 256, 0, stream>>>(folded, w_out, b_out, out);
  stats_kernel<<<64 * B_, 256, 0, stream>>>(out, sums);
  final_kernel<<<NELEM / 256, 256, 0, stream>>>(x, gamma, beta, sums, out);
}
